// CondConv_72782515798874
// MI455X (gfx1250) — matrix operand order repaired and checked by reading, not compiled, not run on hardware
//
#include <hip/hip_runtime.h>
#include <hip/hip_bf16.h>
#include <math.h>

// ---------------------------------------------------------------------------
// CondConv for MI455X (gfx1250, wave32, WMMA).
//   b=32, Cin=Cout=128, H=W=64, K=3, SAME pad, stride 1. 38.7 GFLOP.
//   Path: bf16 WMMA (f32 accum), x pre-transposed to zero-padded NHWC so the
//   conv kernel is pure {global_load_b128 -> v_wmma} with immediate offsets:
//   no LDS, no barriers, no predication. loads/WMMA = 2.5.
// ---------------------------------------------------------------------------

typedef __attribute__((ext_vector_type(16))) __bf16 v16bf;
typedef __attribute__((ext_vector_type(8)))  float  v8f;

#define B_      32
#define CIN     128
#define COUT    128
#define HW      64
#define HWP     66     // padded spatial (zero border of 1)
#define E_      8
#define NPOS    9
#define NCHUNK  4      // 128 channels / 32
#define KC      32     // WMMA bf16 K

// ---------------------------------------------------------------------------
// Kernel 1: routing[b][e] = sigmoid( (1/4096)*sum_c sum_hw x * fcw[e][c] + fcb[e] )
// ---------------------------------------------------------------------------
__global__ void cc_routing(const float* __restrict__ x,
                           const float* __restrict__ fcw,
                           const float* __restrict__ fcb,
                           float* __restrict__ routing) {
    const int b = blockIdx.x;
    const int t = threadIdx.x;           // 256 threads
    float logit[E_];
#pragma unroll
    for (int e = 0; e < E_; ++e) logit[e] = 0.f;

    const float* xb = x + (size_t)b * CIN * HW * HW;
    for (int c = 0; c < CIN; ++c) {
        const float* xc = xb + (size_t)c * HW * HW;
        float p = 0.f;
#pragma unroll
        for (int k = 0; k < 16; ++k)          // 4096/256, coalesced
            p += xc[t + k * 256];
#pragma unroll
        for (int e = 0; e < E_; ++e)
            logit[e] = fmaf(p, fcw[e * CIN + c], logit[e]);
    }

    __shared__ float red[256];
#pragma unroll
    for (int e = 0; e < E_; ++e) {
        red[t] = logit[e];
        __syncthreads();
        for (int s = 128; s > 0; s >>= 1) {
            if (t < s) red[t] += red[t + s];
            __syncthreads();
        }
        if (t == 0) {
            float z = red[0] * (1.0f / (HW * HW)) + fcb[e];
            routing[b * E_ + e] = 1.0f / (1.0f + expf(-z));
        }
        __syncthreads();
    }
}

// ---------------------------------------------------------------------------
// Kernel 2: x (f32, NCHW) -> xt (bf16, zero-padded NHWC [b][66][66][128]).
// One block per (b, padded row). LDS tile makes both sides coalesced.
// ---------------------------------------------------------------------------
__global__ void cc_nhwc(const float* __restrict__ x, __bf16* __restrict__ xt) {
    const int hp = blockIdx.x;           // 0..65
    const int b  = blockIdx.y;
    const int t  = threadIdx.x;          // 256
    __bf16* orow = xt + ((size_t)b * HWP + hp) * HWP * CIN;

    if (hp == 0 || hp == HWP - 1) {      // border row -> zeros
        for (int i = t; i < HWP * CIN; i += 256) orow[i] = (__bf16)0.0f;
        return;
    }

    __shared__ __bf16 ls[HW * CIN];      // [w][c], 16 KB
    const int h = hp - 1;
    const float* xb = x + ((size_t)b * CIN) * HW * HW + (size_t)h * HW;
    for (int i = t; i < CIN * HW; i += 256) {
        const int c = i >> 6, w = i & 63;            // consecutive t -> w
        ls[w * CIN + c] = (__bf16)xb[(size_t)c * HW * HW + w];
    }
    __syncthreads();
    for (int i = t; i < HWP * CIN; i += 256) {
        const int wp = i >> 7, c = i & 127;          // consecutive t -> c
        __bf16 v = (__bf16)0.0f;
        if (wp >= 1 && wp <= HW) v = ls[(wp - 1) * CIN + c];
        orow[i] = v;
    }
}

// ---------------------------------------------------------------------------
// Kernel 3: combine per-sample weights, pre-permuted into WMMA A-fragment
// lane order. wpack[b][pos][ichunk][o][slot]:
//   slots 0..15  = K 0..15   (lanes 0..15)
//   slots 16..31 = K 16..31  (lanes 16..31)
// so lane L reads its 16 A values as one contiguous 32B block.
// ---------------------------------------------------------------------------
__global__ void cc_wpack(const float* __restrict__ routing,
                         const float* __restrict__ kw,   // [E][O][I][3][3]
                         __bf16* __restrict__ wpack) {
    size_t f = (size_t)blockIdx.x * blockDim.x + threadIdx.x;
    const int slot = (int)(f & 31);      f >>= 5;
    const int o    = (int)(f & 127);     f >>= 7;
    const int ic   = (int)(f & 3);       f >>= 2;
    const int pos  = (int)(f % NPOS);
    const int b    = (int)(f / NPOS);

    const int sg = slot >> 3;            // slot-group -> K-group {0,1,2,3}
    const int kbase = sg * 8;
    const int i = ic * KC + kbase + (slot & 7);
    const int ky = pos / 3, kx = pos % 3;

    float acc = 0.f;
#pragma unroll
    for (int e = 0; e < E_; ++e) {
        const size_t kwi = ((((size_t)e * COUT + o) * CIN + i) * 3 + ky) * 3 + kx;
        acc = fmaf(routing[b * E_ + e], kw[kwi], acc);
    }
    const size_t out = (((((size_t)b * NPOS + pos) * NCHUNK + ic) * COUT + o) << 5) + slot;
    wpack[out] = (__bf16)acc;
}

// ---------------------------------------------------------------------------
// Kernel 4: conv as implicit GEMM, direct from padded-NHWC global memory.
// Block = 256 thr (8 waves) -> M=128 (all c_out) x N=64 (one image row).
// Wave `wid` owns rows wid*16..+15, full N=64 -> 4 f32 accumulators.
// All 8 waves load identical B fragments -> served by WGP$.
// Every tap/fragment address is base VGPR + immediate offset.
// ---------------------------------------------------------------------------
__global__ void __launch_bounds__(256, 2)
cc_conv(const __bf16* __restrict__ xt,     // [B][66][66][128] padded NHWC
        const __bf16* __restrict__ wpack,  // [B][9][4][128][32] fragment order
        float* __restrict__ out) {         // [B][128][64][64]
    const int h   = blockIdx.x;            // output row 0..63
    const int b   = blockIdx.y;            // sample
    const int tid = threadIdx.x;
    const int lane = tid & 31;
    const int wid  = tid >> 5;             // 0..7 : 16-row M slice
    const int half = lane >> 4;
    const int l16  = lane & 15;

    v8f acc0 = {}, acc1 = {}, acc2 = {}, acc3 = {};

    const __bf16* xb = xt + (size_t)b * HWP * HWP * CIN;
    const __bf16* wb = wpack + ((size_t)b * NPOS) * NCHUNK * COUT * KC;

    for (int ic = 0; ic < NCHUNK; ++ic) {
        // per-lane bases; everything below is immediate offsets
        const __bf16* bbase = xb + ((size_t)h * HWP + l16) * CIN + ic * KC + half * 16;
        const __bf16* abase = wb + (((size_t)ic * COUT) + wid * 16 + l16) * KC + half * 16;
#pragma unroll
        for (int ky = 0; ky < 3; ++ky) {
#pragma unroll
            for (int kx = 0; kx < 3; ++kx) {
                const int pos = ky * 3 + kx;
                // A: one contiguous 32B load per lane (pos stride = 32768B imm)
                v16bf a = *(const v16bf*)(abase + (size_t)pos * NCHUNK * COUT * KC);
                // B: 4 fragments, offsets (ky*66 + kx + j*16)*256 bytes
                v16bf b0 = *(const v16bf*)(bbase + ((size_t)ky * HWP + kx +  0) * CIN);
                v16bf b1 = *(const v16bf*)(bbase + ((size_t)ky * HWP + kx + 16) * CIN);
                v16bf b2 = *(const v16bf*)(bbase + ((size_t)ky * HWP + kx + 32) * CIN);
                v16bf b3 = *(const v16bf*)(bbase + ((size_t)ky * HWP + kx + 48) * CIN);
                acc0 = __builtin_amdgcn_wmma_f32_16x16x32_bf16(
                           false, a, false, b0, (short)0, acc0, false, false);
                acc1 = __builtin_amdgcn_wmma_f32_16x16x32_bf16(
                           false, a, false, b1, (short)0, acc1, false, false);
                acc2 = __builtin_amdgcn_wmma_f32_16x16x32_bf16(
                           false, a, false, b2, (short)0, acc2, false, false);
                acc3 = __builtin_amdgcn_wmma_f32_16x16x32_bf16(
                           false, a, false, b3, (short)0, acc3, false, false);
            }
        }
    }

    // epilogue: f32 C layout: VGPR r -> M = r + half*8, N = l16
    const int obase = wid * 16 + half * 8;
    float* ob = out + (((size_t)b * COUT) * HW + h) * HW;
#pragma unroll
    for (int r = 0; r < 8; ++r) {
        float* orow = ob + (size_t)(obase + r) * HW * HW + l16;
        orow[ 0] = acc0[r];
        orow[16] = acc1[r];
        orow[32] = acc2[r];
        orow[48] = acc3[r];
    }
}

// ---------------------------------------------------------------------------
extern "C" void kernel_launch(void* const* d_in, const int* in_sizes, int n_in,
                              void* d_out, int out_size, void* d_ws, size_t ws_size,
                              hipStream_t stream) {
    const float* x    = (const float*)d_in[0];   // [32,128,64,64]
    const float* fc_w = (const float*)d_in[1];   // [8,128]
    const float* fc_b = (const float*)d_in[2];   // [8]
    const float* kw   = (const float*)d_in[3];   // [8,128,128,3,3]
    float* out = (float*)d_out;                  // [32,128,64,64]

    char* ws = (char*)d_ws;
    float*  routing = (float*)ws;                                  //   1 KB
    __bf16* wpack   = (__bf16*)(ws + 1024);                        // 9.4 MB
    __bf16* xt      = (__bf16*)(ws + 1024 +
                       (size_t)B_ * NPOS * NCHUNK * COUT * KC * 2);// 35.7 MB

    cc_routing<<<B_, 256, 0, stream>>>(x, fc_w, fc_b, routing);

    dim3 gt(HWP, B_);                                              // hp, b
    cc_nhwc<<<gt, 256, 0, stream>>>(x, xt);

    const size_t nw = (size_t)B_ * NPOS * NCHUNK * COUT * KC;      // 4,718,592
    cc_wpack<<<(unsigned)(nw / 256), 256, 0, stream>>>(routing, kw, wpack);

    dim3 gc(HW, B_);                                               // h, b
    cc_conv<<<gc, 256, 0, stream>>>(xt, wpack, out);
}
